// Qwen3Attention_78769700209091
// MI455X (gfx1250) — matrix operand order repaired and checked by reading, not compiled, not run on hardware
//
#include <hip/hip_runtime.h>
#include <hip/hip_bf16.h>

#define S_   2048
#define HID_ 2048
#define H_   16
#define KH_  8
#define D_   128
#define QKVN 4096   // 2048 q + 1024 k + 1024 v
#define GK   2048   // contraction dim of both big GEMMs (compile-time for offset folding)

typedef __attribute__((ext_vector_type(16))) _Float16 v16h;
typedef __attribute__((ext_vector_type(8)))  _Float16 v8h;
typedef __attribute__((ext_vector_type(4)))  _Float16 v4h;
typedef __attribute__((ext_vector_type(8)))  float    v8f;

__device__ __forceinline__ v16h cat16(v8h lo, v8h hi) {
  return __builtin_shufflevector(lo, hi, 0,1,2,3,4,5,6,7,8,9,10,11,12,13,14,15);
}

__device__ __forceinline__ v8f wmma_f16(v16h a, v16h b, v8f c) {
  return __builtin_amdgcn_wmma_f32_16x16x32_f16(false, a, false, b, (short)0, c, false, false);
}

// ---------------- f32 -> f16 convert (vectorized x4) ----------------
__global__ void conv_f16(const float* __restrict__ src, _Float16* __restrict__ dst, int n) {
  int i = 4 * (blockIdx.x * blockDim.x + threadIdx.x);
  if (i >= n) return;
  float4 v = *(const float4*)(src + i);
  v4h o; o[0] = (_Float16)v.x; o[1] = (_Float16)v.y; o[2] = (_Float16)v.z; o[3] = (_Float16)v.w;
  *(v4h*)(dst + i) = o;
}

// ---------------- WMMA GEMM: C[M,N] = A[M,GK] * B[N,GK]^T ----------------
// One wave computes a 64x64 tile: 16 v_wmma per K=32 step (32 FLOP/byte).
// Sub-tile row offsets (i*16*GK) are compile-time -> fold into vmem offsets.
__global__ void __launch_bounds__(128) gemm_f16_wmma(const _Float16* __restrict__ A,
                                                     const _Float16* __restrict__ B,
                                                     float* __restrict__ C,
                                                     int M, int N) {
  int wave = (blockIdx.x * blockDim.x + threadIdx.x) >> 5;
  int lane = threadIdx.x & 31;
  int tn = N >> 6, tm = M >> 6;
  if (wave >= tm * tn) return;
  int ti = wave / tn, tj = wave % tn;
  int m = lane & 15, g = lane >> 4;

  v8f z = {};
  v8f acc[4][4];
#pragma unroll
  for (int i = 0; i < 4; ++i)
#pragma unroll
    for (int j = 0; j < 4; ++j) acc[i][j] = z;

  const _Float16* ap = A + (size_t)(ti * 64 + m) * GK;   // + i*16*GK folded as const
  const _Float16* bp = B + (size_t)(tj * 64 + m) * GK;   // + j*16*GK folded as const

  for (int kk = 0; kk < GK; kk += 32) {
    // prefetch 4 K-steps ahead (lowers to global_prefetch_b8)
    __builtin_prefetch(ap + kk + 128, 0, 1);
    __builtin_prefetch(bp + kk + 128, 0, 1);
    v16h a[4];
#pragma unroll
    for (int i = 0; i < 4; ++i)
      a[i] = *(const v16h*)(ap + i * 16 * GK + kk + 16 * g);
#pragma unroll
    for (int j = 0; j < 4; ++j) {
      v16h b = *(const v16h*)(bp + j * 16 * GK + kk + 16 * g);
#pragma unroll
      for (int i = 0; i < 4; ++i) acc[i][j] = wmma_f16(a[i], b, acc[i][j]);
    }
  }

#pragma unroll
  for (int i = 0; i < 4; ++i)
#pragma unroll
    for (int r = 0; r < 8; ++r) {
      size_t row = (size_t)(ti * 64 + i * 16 + 8 * g + r) * N + tj * 64;
#pragma unroll
      for (int j = 0; j < 4; ++j) C[row + j * 16 + m] = acc[i][j][r];
    }
}

// ---------------- RMSNorm + RoPE; writes q[h][s][d], k[kvh][s][d], v^T[kvh][d][s] ----------------
__global__ void __launch_bounds__(128) normrope(const float* __restrict__ QKV,
                                                const float* __restrict__ qw,
                                                const float* __restrict__ kw,
                                                const float* __restrict__ cosT,
                                                const float* __restrict__ sinT,
                                                _Float16* __restrict__ q_h,
                                                _Float16* __restrict__ k_h,
                                                _Float16* __restrict__ v_t) {
  int wave = (blockIdx.x * blockDim.x + threadIdx.x) >> 5;
  int lane = threadIdx.x & 31;
  int slot = wave & 31;
  int s    = wave >> 5;
  if (s >= S_) return;
  int d0 = lane * 4;

  if (slot >= 24) {  // V: convert + transpose
    int h = slot - 24;
    float4 v = *(const float4*)(QKV + (size_t)s * QKVN + 3072 + h * 128 + d0);
    _Float16* dst = v_t + ((size_t)h * D_ + d0) * S_ + s;
    dst[0]      = (_Float16)v.x;
    dst[S_]     = (_Float16)v.y;
    dst[2 * S_] = (_Float16)v.z;
    dst[3 * S_] = (_Float16)v.w;
    return;
  }
  bool isq = slot < 16;
  int h = isq ? slot : slot - 16;
  const float* p = QKV + (size_t)s * QKVN + (isq ? h * 128 : 2048 + h * 128) + d0;
  float4 t = *(const float4*)p;

  float ss = t.x * t.x + t.y * t.y + t.z * t.z + t.w * t.w;
#pragma unroll
  for (int off = 16; off >= 1; off >>= 1) ss += __shfl_xor(ss, off, 32);
  float rinv = rsqrtf(ss * (1.0f / 128.0f) + 1e-6f);

  const float* w = isq ? qw : kw;
  float4 wv = *(const float4*)(w + d0);
  float n0 = t.x * rinv * wv.x, n1 = t.y * rinv * wv.y;
  float n2 = t.z * rinv * wv.z, n3 = t.w * rinv * wv.w;

  // rotate_half partner lives at lane ^ 16 (d +/- 64)
  float o0 = __shfl_xor(n0, 16, 32), o1 = __shfl_xor(n1, 16, 32);
  float o2 = __shfl_xor(n2, 16, 32), o3 = __shfl_xor(n3, 16, 32);
  float sgn = (lane < 16) ? -1.0f : 1.0f;

  float4 c  = *(const float4*)(cosT + (size_t)s * D_ + d0);
  float4 sv = *(const float4*)(sinT + (size_t)s * D_ + d0);
  float r0 = n0 * c.x + sgn * o0 * sv.x;
  float r1 = n1 * c.y + sgn * o1 * sv.y;
  float r2 = n2 * c.z + sgn * o2 * sv.z;
  float r3 = n3 * c.w + sgn * o3 * sv.w;

  _Float16* dst = (isq ? q_h : k_h) + ((size_t)h * S_ + s) * D_ + d0;
  v4h o; o[0] = (_Float16)r0; o[1] = (_Float16)r1; o[2] = (_Float16)r2; o[3] = (_Float16)r3;
  *(v4h*)dst = o;
}

// ---------------- flash attention: one wave per (head, 16-query block) ----------------
__global__ void __launch_bounds__(128) attn_kernel(const _Float16* __restrict__ q_h,
                                                   const _Float16* __restrict__ k_h,
                                                   const _Float16* __restrict__ v_t,
                                                   _Float16* __restrict__ o_h) {
  __shared__ __align__(32) _Float16 plds[4][16 * 32];
  int wave = (blockIdx.x * blockDim.x + threadIdx.x) >> 5;
  int lane = threadIdx.x & 31;
  int wl   = (threadIdx.x >> 5) & 3;
  int h  = wave >> 7;       // 128 query blocks per head
  int qb = wave & 127;
  int kvh = h >> 1;         // GQA: groups = H/KH = 2
  int m = lane & 15, g = lane >> 4, n = m;

  // resident Q: 16x128 as four 16x32 A-tiles
  const _Float16* qp = q_h + ((size_t)h * S_ + qb * 16 + m) * D_;
  v16h aq[4];
#pragma unroll
  for (int t = 0; t < 4; ++t)
    aq[t] = *(const v16h*)(qp + 32 * t + 16 * g);

  v8f z = {};
  v8f acc[8];
#pragma unroll
  for (int t = 0; t < 8; ++t) acc[t] = z;
  float mrow[8], lrow[8];
#pragma unroll
  for (int r = 0; r < 8; ++r) { mrow[r] = -3.0e38f; lrow[r] = 0.0f; }

  const float scale = 0.08838834764831845f;  // 1/sqrt(128)
  int nkb = (qb >> 1) + 1;
  _Float16* pl = plds[wl];

  for (int kb = 0; kb < nkb; ++kb) {
    int k0 = kb * 32;
    v8f c0 = z, c1 = z;
    const _Float16* kp0 = k_h + ((size_t)kvh * S_ + k0 + n) * D_;
    const _Float16* kp1 = kp0 + (size_t)16 * D_;
#pragma unroll
    for (int t = 0; t < 4; ++t) c0 = wmma_f16(aq[t], *(const v16h*)(kp0 + 32 * t + 16 * g), c0);
#pragma unroll
    for (int t = 0; t < 4; ++t) c1 = wmma_f16(aq[t], *(const v16h*)(kp1 + 32 * t + 16 * g), c1);

#pragma unroll
    for (int r = 0; r < 8; ++r) {
      int q = qb * 16 + 8 * g + r;
      float s0 = c0[r] * scale; bool ok0 = (k0 + n) <= q;
      float s1 = c1[r] * scale; bool ok1 = (k0 + 16 + n) <= q;
      float sm = fmaxf(ok0 ? s0 : -3.0e38f, ok1 ? s1 : -3.0e38f);
#pragma unroll
      for (int off = 8; off >= 1; off >>= 1) sm = fmaxf(sm, __shfl_xor(sm, off, 32));
      float mnew = fmaxf(mrow[r], sm);
      float fac  = __expf(mrow[r] - mnew);
      float p0 = ok0 ? __expf(s0 - mnew) : 0.0f;
      float p1 = ok1 ? __expf(s1 - mnew) : 0.0f;
      float rs = p0 + p1;
#pragma unroll
      for (int off = 8; off >= 1; off >>= 1) rs += __shfl_xor(rs, off, 32);
      lrow[r] = lrow[r] * fac + rs;
      mrow[r] = mnew;
#pragma unroll
      for (int t = 0; t < 8; ++t) acc[t][r] *= fac;
      pl[(8 * g + r) * 32 + n]      = (_Float16)p0;
      pl[(8 * g + r) * 32 + 16 + n] = (_Float16)p1;
    }
    asm volatile("s_wait_dscnt 0" ::: "memory");  // wave-internal LDS RAW (C-layout -> A-layout)
    v16h ap = *(const v16h*)(pl + m * 32 + 16 * g);

    const _Float16* vp = v_t + ((size_t)kvh * D_ + n) * S_ + k0 + 16 * g;
#pragma unroll
    for (int t = 0; t < 8; ++t)
      acc[t] = wmma_f16(ap, *(const v16h*)(vp + (size_t)t * 16 * S_), acc[t]);
  }

#pragma unroll
  for (int r = 0; r < 8; ++r) {
    float inv = 1.0f / lrow[r];
    size_t row = (size_t)(qb * 16 + 8 * g + r) * (H_ * D_) + (size_t)h * D_;
#pragma unroll
    for (int t = 0; t < 8; ++t) o_h[row + t * 16 + n] = (_Float16)(acc[t][r] * inv);
  }
}

extern "C" void kernel_launch(void* const* d_in, const int* in_sizes, int n_in,
                              void* d_out, int out_size, void* d_ws, size_t ws_size,
                              hipStream_t stream) {
  const float* x    = (const float*)d_in[0];
  const float* Wq   = (const float*)d_in[1];
  const float* Wk   = (const float*)d_in[2];
  const float* Wv   = (const float*)d_in[3];
  const float* Wo   = (const float*)d_in[4];
  const float* qw   = (const float*)d_in[5];
  const float* kw   = (const float*)d_in[6];
  const float* cosT = (const float*)d_in[7];
  const float* sinT = (const float*)d_in[8];
  float* out = (float*)d_out;

  char* ws = (char*)d_ws;
  const size_t MB = 1024ull * 1024ull;
  _Float16* x_h  = (_Float16*)(ws + 0 * MB);    // 2048x2048 f16      (8 MB)
  _Float16* W_h  = (_Float16*)(ws + 8 * MB);    // 4096x2048 f16      (16 MB)  [Wq;Wk;Wv]
  _Float16* Wo_h = (_Float16*)(ws + 24 * MB);   // 2048x2048 f16      (8 MB)
  float*    QKV  = (float*)   (ws + 32 * MB);   // 2048x4096 f32      (32 MB)
  _Float16* q_h  = (_Float16*)(ws + 64 * MB);   // [16][2048][128]    (8 MB)
  _Float16* k_h  = (_Float16*)(ws + 72 * MB);   // [8][2048][128]     (4 MB)
  _Float16* v_t  = (_Float16*)(ws + 76 * MB);   // [8][128][2048]     (4 MB)
  _Float16* o_h  = (_Float16*)(ws + 80 * MB);   // 2048x2048 f16      (8 MB)

  // 1) converts
  conv_f16<<<4096, 256, 0, stream>>>(x,  x_h,               HID_ * S_);
  conv_f16<<<4096, 256, 0, stream>>>(Wq, W_h,               2048 * 2048);
  conv_f16<<<2048, 256, 0, stream>>>(Wk, W_h + 4194304,     1024 * 2048);
  conv_f16<<<2048, 256, 0, stream>>>(Wv, W_h + 6291456,     1024 * 2048);
  conv_f16<<<4096, 256, 0, stream>>>(Wo, Wo_h,              2048 * 2048);

  // 2) fused QKV projection: [2048,4096] = x_h[2048,2048] @ W_h[4096,2048]^T
  //    waves = (2048/64)*(4096/64) = 2048 -> 512 blocks of 128
  gemm_f16_wmma<<<512, 128, 0, stream>>>(x_h, W_h, QKV, S_, QKVN);

  // 3) RMSNorm + RoPE (+ V transpose)
  normrope<<<16384, 128, 0, stream>>>(QKV, qw, kw, cosT, sinT, q_h, k_h, v_t);

  // 4) causal GQA flash attention
  attn_kernel<<<512, 128, 0, stream>>>(q_h, k_h, v_t, o_h);

  // 5) output projection: out[2048,2048] = o_h @ Wo^T
  //    waves = (2048/64)*(2048/64) = 1024 -> 256 blocks of 128
  gemm_f16_wmma<<<256, 128, 0, stream>>>(o_h, Wo_h, out, S_, HID_);
}
